// temporalAttention_45938970198782
// MI455X (gfx1250) — compile-verified, matrix operand order checked by reading, not run on hardware
//
#include <hip/hip_runtime.h>

// ---------------------------------------------------------------------------
// CDNA5 (gfx1250) fused temporal-attention block.
// All matrix math runs on v_wmma_f32_16x16x32_bf16 (wave32 WMMA).
// Activations row-major [token][D] bf16; weights pre-packed transposed [N][K]
// bf16 so every WMMA fragment is two contiguous 16B loads per lane.
// GEMM waves own 2 M-tiles x 4 N-tiles so each B fragment feeds two
// back-to-back identical WMMAs (reuse_b hint legal per ISA 7.12).
// ---------------------------------------------------------------------------

typedef __bf16 bf16;
typedef __attribute__((ext_vector_type(16))) __bf16 v16bf;
typedef __attribute__((ext_vector_type(8)))  float  v8f;

#define DEV __device__ __forceinline__

static constexpr int Bsz  = 4;
static constexpr int Tsz  = 64;
static constexpr int Nsz  = 256;
static constexpr int Dm   = 512;
static constexpr int HEADS = 8;
static constexpr int HD   = 64;                    // head dim
static constexpr int MTOK = Bsz * Tsz * Nsz;       // 65536 tokens

// ---------- bf16 <-> f32 helpers (bit-exact RNE, no reliance on __bf16 ops)
DEV bf16 f2bf(float f) {
  union { float f; unsigned u; } in; in.f = f;
  unsigned r = (in.u + 0x7fffu + ((in.u >> 16) & 1u)) >> 16;
  union { unsigned short s; bf16 b; } out; out.s = (unsigned short)r;
  return out.b;
}
DEV float bf2f(bf16 b) {
  union { unsigned short s; bf16 b; } in; in.b = b;
  union { unsigned u; float f; } out; out.u = ((unsigned)in.s) << 16;
  return out.f;
}

// ---------- fragment loads ------------------------------------------------
// 16-bit A (16x32) / B (32x16) layout: lane L -> row/col = L&15, h = L>>4.
// Lane's 16 elements = K runs [h*8, h*8+7] and [16+h*8, 16+h*8+7]:
// two contiguous 16-byte chunks of the row-major operand.
DEV v16bf load_frag(const bf16* __restrict__ base, int ldr, int row, int k0) {
  const int lane = threadIdx.x & 31;
  const int h = (lane >> 4) & 1;
  const bf16* p = base + (size_t)row * (size_t)ldr + (size_t)(k0 + h * 8);
  union { uint4 u[2]; v16bf v; } f;
  f.u[0] = *reinterpret_cast<const uint4*>(p);
  f.u[1] = *reinterpret_cast<const uint4*>(p + 16);
  return f.v;
}
DEV v16bf load_frag_pred(const bf16* __restrict__ base, int ldr, int row, int k0,
                         bool valid) {
  union { uint4 u[2]; v16bf v; } f;
  if (valid) {
    const int lane = threadIdx.x & 31;
    const int h = (lane >> 4) & 1;
    const bf16* p = base + (size_t)row * (size_t)ldr + (size_t)(k0 + h * 8);
    f.u[0] = *reinterpret_cast<const uint4*>(p);
    f.u[1] = *reinterpret_cast<const uint4*>(p + 16);
  } else {
    f.u[0] = make_uint4(0u, 0u, 0u, 0u);
    f.u[1] = make_uint4(0u, 0u, 0u, 0u);
  }
  return f.v;
}
DEV v8f wmma_bf16(v16bf a, v16bf b, v8f c) {
  return __builtin_amdgcn_wmma_f32_16x16x32_bf16(
      /*neg_a=*/false, a, /*neg_b=*/false, b,
      /*c_mod=*/(short)0, c, /*reuse_a=*/false, /*reuse_b=*/false);
}
// Second of a back-to-back identical pair consuming the same B operand.
DEV v8f wmma_bf16_rb(v16bf a, v16bf b, v8f c) {
  return __builtin_amdgcn_wmma_f32_16x16x32_bf16(
      /*neg_a=*/false, a, /*neg_b=*/false, b,
      /*c_mod=*/(short)0, c, /*reuse_a=*/false, /*reuse_b=*/true);
}
DEV v8f zero8() {
  v8f z;
#pragma unroll
  for (int i = 0; i < 8; ++i) z[i] = 0.f;
  return z;
}

// ---------- C/D tile store (lane: col = lane&15, rows = r + (lane>>4)*8) ---
// MODE 0: bf16 row-major; MODE 1: f32 row-major; MODE 2: V^T per-head scatter
template <int MODE, bool RELU, bool RESID>
DEV void store_tile(const v8f& c, int mrow0, int ncol, int Ncols,
                    const float* __restrict__ bias,
                    const float* __restrict__ resid, void* __restrict__ out) {
  const float bv = bias[ncol];
  #pragma unroll
  for (int r = 0; r < 8; ++r) {
    const int mrow = mrow0 + r;
    float v = c[r] + bv;
    if (RELU) v = v > 0.f ? v : 0.f;
    if (RESID) v += resid[(size_t)mrow * Ncols + ncol];
    if (MODE == 0) {
      ((bf16*)out)[(size_t)mrow * Ncols + ncol] = f2bf(v);
    } else if (MODE == 1) {
      ((float*)out)[(size_t)mrow * Ncols + ncol] = v;
    } else {  // scatter V into [head_global][dv][t], head_global=(b*256+n)*8+h
      const int n = mrow & 255, bt = mrow >> 8, t = bt & 63, b = bt >> 6;
      const int hd = ncol >> 6, dv = ncol & 63;
      const size_t idx =
          ((((size_t)((b << 8) + n) * HEADS + hd) * HD) + dv) * Tsz + t;
      ((bf16*)out)[idx] = f2bf(v);
    }
  }
}

// ---------- generic streaming GEMM: C = A[MxK]*Bpk[N][K]^T + bias ----------
// block = 8 waves; wave tile = 2 M-strips (16 rows, 128 apart) x 64 cols;
// block tile = 256 x 64.  Grid: (M/256, N/64).
// Each B fragment feeds two consecutive identical WMMAs -> reuse_b hint.
template <int MODE, bool RELU, bool RESID>
__global__ __launch_bounds__(256) void gemm_bf16(
    const bf16* __restrict__ A, const bf16* __restrict__ Bpk,
    const float* __restrict__ bias, const float* __restrict__ resid,
    void* __restrict__ out, int Kdim, int Ncols) {
  const int wave = threadIdx.x >> 5;
  const int lane = threadIdx.x & 31;
  const int l15 = lane & 15;
  const int h8 = ((lane >> 4) & 1) * 8;
  const int mWave = blockIdx.x * 256 + wave * 16;  // strip 0; strip 1 = +128
  const int nBlk = blockIdx.y * 64;

  v8f acc[2][4];
  #pragma unroll
  for (int ms = 0; ms < 2; ++ms)
    #pragma unroll
    for (int i = 0; i < 4; ++i) acc[ms][i] = zero8();

  const int arow0 = mWave + l15;
  const int arow1 = arow0 + 128;
  for (int kk = 0; kk < Kdim; kk += 32) {
    __builtin_prefetch(A + (size_t)arow0 * Kdim + kk + 64, 0, 1);
    __builtin_prefetch(A + (size_t)arow1 * Kdim + kk + 64, 0, 1);
    const v16bf a0 = load_frag(A, Kdim, arow0, kk);
    const v16bf a1 = load_frag(A, Kdim, arow1, kk);
    #pragma unroll
    for (int nt = 0; nt < 4; ++nt) {
      const v16bf b = load_frag(Bpk, Kdim, nBlk + nt * 16 + l15, kk);
      acc[0][nt] = wmma_bf16(a0, b, acc[0][nt]);
      acc[1][nt] = wmma_bf16_rb(a1, b, acc[1][nt]);  // same B, reuse hint
    }
  }
  #pragma unroll
  for (int ms = 0; ms < 2; ++ms)
    #pragma unroll
    for (int nt = 0; nt < 4; ++nt)
      store_tile<MODE, RELU, RESID>(acc[ms][nt], mWave + ms * 128 + h8,
                                    nBlk + nt * 16 + l15, Ncols, bias, resid,
                                    out);
}

// ---------- conv along N (kernel 3) as 3 shifted GEMMs ---------------------
__global__ __launch_bounds__(256) void conv_gemm(
    const bf16* __restrict__ A, const bf16* __restrict__ Wpk3,
    const float* __restrict__ bias, bf16* __restrict__ out) {
  const int wave = threadIdx.x >> 5;
  const int lane = threadIdx.x & 31;
  const int l15 = lane & 15;
  const int h8 = ((lane >> 4) & 1) * 8;
  const int mWave = blockIdx.x * 128 + wave * 16;
  const int nBlk = blockIdx.y * 64;

  v8f acc[4];
  #pragma unroll
  for (int i = 0; i < 4; ++i) acc[i] = zero8();

  const int rowL = mWave + l15;
  const int nIdx = rowL & (Nsz - 1);
  for (int j = 0; j < 3; ++j) {
    const bf16* Bpk = Wpk3 + (size_t)j * Dm * Dm;
    const int sh = j - 1;
    const bool valid = ((unsigned)(nIdx + sh)) < (unsigned)Nsz;
    const int arow = rowL + sh;
    for (int kk = 0; kk < Dm; kk += 32) {
      const v16bf a = load_frag_pred(A, Dm, arow, kk, valid);
      #pragma unroll
      for (int nt = 0; nt < 4; ++nt) {
        const v16bf b = load_frag(Bpk, Dm, nBlk + nt * 16 + l15, kk);
        acc[nt] = wmma_bf16(a, b, acc[nt]);
      }
    }
  }
  #pragma unroll
  for (int nt = 0; nt < 4; ++nt)
    store_tile<0, false, false>(acc[nt], mWave + h8, nBlk + nt * 16 + l15,
                                Dm, bias, nullptr, (void*)out);
}

// ---------- attention: one wave per (b,n,head); T=64, d=64 -----------------
__global__ __launch_bounds__(128) void attention_kernel(
    const bf16* __restrict__ Q, const bf16* __restrict__ Kx,
    const bf16* __restrict__ Vt, bf16* __restrict__ Og) {
  __shared__ bf16 Pl[4 * Tsz * Tsz];  // 4 waves x 8KB
  const int wave = threadIdx.x >> 5;
  const int lane = threadIdx.x & 31;
  const int l15 = lane & 15;
  const int h8 = ((lane >> 4) & 1) * 8;
  const int head_global = blockIdx.x * 4 + wave;  // (b*256+n)*8 + h
  const int h = head_global & (HEADS - 1);
  const int bn = head_global >> 3;
  const int n = bn & (Nsz - 1);
  const int b = bn >> 8;

  const int ldTok = Nsz * Dm;  // stride between consecutive t
  const bf16* Qb = Q + ((size_t)b * Tsz * Nsz + n) * Dm + h * HD;
  const bf16* Kb = Kx + ((size_t)b * Tsz * Nsz + n) * Dm + h * HD;

  // ---- S = Q K^T  (both row-major [t][d]; B frag gathers rows of K) ----
  v8f S[4][4];
  #pragma unroll
  for (int i = 0; i < 4; ++i)
    #pragma unroll
    for (int j = 0; j < 4; ++j) S[i][j] = zero8();

  for (int kk = 0; kk < HD; kk += 32) {
    v16bf aq[4], bk[4];
    #pragma unroll
    for (int mt = 0; mt < 4; ++mt) aq[mt] = load_frag(Qb, ldTok, mt * 16 + l15, kk);
    #pragma unroll
    for (int nt = 0; nt < 4; ++nt) bk[nt] = load_frag(Kb, ldTok, nt * 16 + l15, kk);
    #pragma unroll
    for (int mt = 0; mt < 4; ++mt)
      #pragma unroll
      for (int nt = 0; nt < 4; ++nt) S[mt][nt] = wmma_bf16(aq[mt], bk[nt], S[mt][nt]);
  }

  // ---- softmax over s (row m lives in one 16-lane half; 4 vals/lane) ----
  const float scale = 0.125f;  // 1/sqrt(64)
  bf16* Pw = Pl + wave * Tsz * Tsz;
  #pragma unroll
  for (int mt = 0; mt < 4; ++mt) {
    #pragma unroll
    for (int r = 0; r < 8; ++r) {
      float mx = -3.4e38f;
      #pragma unroll
      for (int nt = 0; nt < 4; ++nt) mx = fmaxf(mx, S[mt][nt][r] * scale);
      #pragma unroll
      for (int msk = 1; msk < 16; msk <<= 1)
        mx = fmaxf(mx, __shfl_xor(mx, msk, 32));
      float sum = 0.f;
      #pragma unroll
      for (int nt = 0; nt < 4; ++nt) {
        const float p = __expf(S[mt][nt][r] * scale - mx);
        S[mt][nt][r] = p;
        sum += p;
      }
      #pragma unroll
      for (int msk = 1; msk < 16; msk <<= 1) sum += __shfl_xor(sum, msk, 32);
      const float rinv = 1.f / sum;
      const int mrow = mt * 16 + r + h8;
      #pragma unroll
      for (int nt = 0; nt < 4; ++nt)
        Pw[mrow * Tsz + nt * 16 + l15] = f2bf(S[mt][nt][r] * rinv);
    }
  }
  __syncthreads();

  // ---- O = P V  (P row-major in LDS; Vt is [head][dv][t]) ----
  const bf16* Vh = Vt + (size_t)head_global * HD * Tsz;
  v8f O[4][4];
  #pragma unroll
  for (int i = 0; i < 4; ++i)
    #pragma unroll
    for (int j = 0; j < 4; ++j) O[i][j] = zero8();

  for (int kk = 0; kk < Tsz; kk += 32) {
    v16bf ap[4], bv[4];
    #pragma unroll
    for (int mt = 0; mt < 4; ++mt) ap[mt] = load_frag(Pw, Tsz, mt * 16 + l15, kk);
    #pragma unroll
    for (int nt = 0; nt < 4; ++nt) bv[nt] = load_frag(Vh, Tsz, nt * 16 + l15, kk);
    #pragma unroll
    for (int mt = 0; mt < 4; ++mt)
      #pragma unroll
      for (int nt = 0; nt < 4; ++nt) O[mt][nt] = wmma_bf16(ap[mt], bv[nt], O[mt][nt]);
  }

  // scatter O back to token-major [tok][D]
  #pragma unroll
  for (int mt = 0; mt < 4; ++mt)
    #pragma unroll
    for (int nt = 0; nt < 4; ++nt) {
      const int dv = nt * 16 + l15;
      #pragma unroll
      for (int r = 0; r < 8; ++r) {
        const int t = mt * 16 + r + h8;
        const size_t tok = ((size_t)b * Tsz + t) * Nsz + n;
        Og[tok * Dm + h * HD + dv] = f2bf(O[mt][nt][r]);
      }
    }
}

// ---------- layernorm (row of 512) -> bf16 --------------------------------
__global__ __launch_bounds__(256) void layernorm_bf16(
    const float* __restrict__ x, const float* __restrict__ g,
    const float* __restrict__ be, bf16* __restrict__ out) {
  const int row = blockIdx.x;
  const float* xr = x + (size_t)row * Dm;
  const float v0 = xr[threadIdx.x];
  const float v1 = xr[threadIdx.x + 256];
  float s = v0 + v1, sq = v0 * v0 + v1 * v1;
  #pragma unroll
  for (int m = 16; m >= 1; m >>= 1) {
    s += __shfl_xor(s, m, 32);
    sq += __shfl_xor(sq, m, 32);
  }
  __shared__ float red[2][8];
  __shared__ float mu_s, rv_s;
  if ((threadIdx.x & 31) == 0) {
    red[0][threadIdx.x >> 5] = s;
    red[1][threadIdx.x >> 5] = sq;
  }
  __syncthreads();
  if (threadIdx.x == 0) {
    float S = 0.f, Qs = 0.f;
    #pragma unroll
    for (int w = 0; w < 8; ++w) { S += red[0][w]; Qs += red[1][w]; }
    const float mu = S * (1.f / Dm);
    const float var = Qs * (1.f / Dm) - mu * mu;
    mu_s = mu;
    rv_s = rsqrtf(var + 1e-5f);
  }
  __syncthreads();
  const float mu = mu_s, rv = rv_s;
  out[(size_t)row * Dm + threadIdx.x] =
      f2bf((v0 - mu) * rv * g[threadIdx.x] + be[threadIdx.x]);
  out[(size_t)row * Dm + threadIdx.x + 256] =
      f2bf((v1 - mu) * rv * g[threadIdx.x + 256] + be[threadIdx.x + 256]);
}

// ---------- TE injection: q += TE@wq_ht + bq_ht (and same for k) ----------
__global__ __launch_bounds__(256) void te_inject(
    bf16* __restrict__ q, bf16* __restrict__ k, const float* __restrict__ TE,
    const float* __restrict__ wqh, const float* __restrict__ bqh,
    const float* __restrict__ wkh, const float* __restrict__ bkh) {
  const size_t i = (size_t)blockIdx.x * 256 + threadIdx.x;  // over M*D
  const int c = (int)(i & (Dm - 1));
  const size_t tok = i >> 9;
  const float te0 = TE[tok * 2], te1 = TE[tok * 2 + 1];
  q[i] = f2bf(bf2f(q[i]) + te0 * wqh[c] + te1 * wqh[Dm + c] + bqh[c]);
  k[i] = f2bf(bf2f(k[i]) + te0 * wkh[c] + te1 * wkh[Dm + c] + bkh[c]);
}

// ---------- weight pack: f32 [K][N] -> bf16 [N][K] ------------------------
__global__ __launch_bounds__(256) void pack_wT(const float* __restrict__ src,
                                               bf16* __restrict__ dst, int K,
                                               int N) {
  const size_t i = (size_t)blockIdx.x * 256 + threadIdx.x;
  if (i >= (size_t)K * N) return;
  const int k = (int)(i / N), nn = (int)(i % N);
  dst[(size_t)nn * K + k] = f2bf(src[i]);
}

// ===========================================================================
extern "C" void kernel_launch(void* const* d_in, const int* in_sizes, int n_in,
                              void* d_out, int out_size, void* d_ws,
                              size_t ws_size, hipStream_t stream) {
  const float* x    = (const float*)d_in[0];
  const float* TE   = (const float*)d_in[1];
  const float* wq   = (const float*)d_in[2];
  const float* bq   = (const float*)d_in[3];
  const float* wk   = (const float*)d_in[4];
  const float* bk   = (const float*)d_in[5];
  const float* wv   = (const float*)d_in[6];
  const float* bv   = (const float*)d_in[7];
  const float* wq_ht = (const float*)d_in[8];
  const float* bq_ht = (const float*)d_in[9];
  const float* wk_ht = (const float*)d_in[10];
  const float* bk_ht = (const float*)d_in[11];
  const float* c1w  = (const float*)d_in[12];
  const float* c1b  = (const float*)d_in[13];
  const float* c2w  = (const float*)d_in[14];
  const float* c2b  = (const float*)d_in[15];
  const float* g1   = (const float*)d_in[16];
  const float* be1  = (const float*)d_in[17];
  const float* g2   = (const float*)d_in[18];
  const float* be2  = (const float*)d_in[19];
  const float* wo   = (const float*)d_in[20];
  const float* bo   = (const float*)d_in[21];
  const float* wf1  = (const float*)d_in[22];
  const float* bf1  = (const float*)d_in[23];
  const float* wf2  = (const float*)d_in[24];
  const float* bf2  = (const float*)d_in[25];

  // workspace carve-up
  const size_t MD = (size_t)MTOK * Dm;
  char* wsb = (char*)d_ws;
  bf16* Axn = (bf16*)wsb;                 wsb += MD * 2;  // xn / h (bf16)
  bf16* Bq  = (bf16*)wsb;                 wsb += MD * 2;  // q ping / ffn hidden[0]
  bf16* Cq  = (bf16*)wsb;                 wsb += MD * 2;  // q pong / ffn hidden[1]
  bf16* Dk  = (bf16*)wsb;                 wsb += MD * 2;  // k ping / ffn hidden[2]
  bf16* Ek  = (bf16*)wsb;                 wsb += MD * 2;  // k pong / ffn hidden[3]
  bf16* Vt  = (bf16*)wsb;                 wsb += MD * 2;  // V^T per head
  bf16* G   = (bf16*)wsb;                 wsb += MD * 2;  // attention output
  float* H  = (float*)wsb;                wsb += MD * 4;  // x1 (f32)
  bf16* pwq = (bf16*)wsb;
  bf16* pwk = pwq + (size_t)Dm * Dm;
  bf16* pwv = pwk + (size_t)Dm * Dm;
  bf16* pc1 = pwv + (size_t)Dm * Dm;                // 3 slices
  bf16* pc2 = pc1 + 3 * (size_t)Dm * Dm;            // 3 slices
  bf16* pwo = pc2 + 3 * (size_t)Dm * Dm;
  bf16* pf1 = pwo + (size_t)Dm * Dm;                // [2048][512]
  bf16* pf2 = pf1 + (size_t)Dm * 4 * Dm;            // [512][2048]
  bf16* hid = Bq;                                   // M x 2048 bf16 (reuse)

  const int packBlocks512 = (Dm * Dm + 255) / 256;
  const int packBlocksF   = (Dm * 4 * Dm + 255) / 256;
  // ---- pack all weights transposed to bf16 ----
  pack_wT<<<packBlocks512, 256, 0, stream>>>(wq, pwq, Dm, Dm);
  pack_wT<<<packBlocks512, 256, 0, stream>>>(wk, pwk, Dm, Dm);
  pack_wT<<<packBlocks512, 256, 0, stream>>>(wv, pwv, Dm, Dm);
  for (int j = 0; j < 3; ++j) {
    pack_wT<<<packBlocks512, 256, 0, stream>>>(c1w + (size_t)j * Dm * Dm,
                                               pc1 + (size_t)j * Dm * Dm, Dm, Dm);
    pack_wT<<<packBlocks512, 256, 0, stream>>>(c2w + (size_t)j * Dm * Dm,
                                               pc2 + (size_t)j * Dm * Dm, Dm, Dm);
  }
  pack_wT<<<packBlocks512, 256, 0, stream>>>(wo, pwo, Dm, Dm);
  pack_wT<<<packBlocksF, 256, 0, stream>>>(wf1, pf1, Dm, 4 * Dm);
  pack_wT<<<packBlocksF, 256, 0, stream>>>(wf2, pf2, 4 * Dm, Dm);

  const dim3 gemmGridD(MTOK / 256, Dm / 64);       // (256, 8)
  const dim3 gemmGridF1(MTOK / 256, 4 * Dm / 64);  // (256, 32)
  const dim3 convGrid(MTOK / 128, Dm / 64);        // (512, 8)

  // ---- LN1: x -> xn (bf16) ----
  layernorm_bf16<<<MTOK, 256, 0, stream>>>(x, g1, be1, Axn);

  // ---- QKV GEMMs ----
  gemm_bf16<0, false, false><<<gemmGridD, 256, 0, stream>>>(
      Axn, pwq, bq, nullptr, (void*)Bq, Dm, Dm);
  gemm_bf16<0, false, false><<<gemmGridD, 256, 0, stream>>>(
      Axn, pwk, bk, nullptr, (void*)Dk, Dm, Dm);
  gemm_bf16<2, false, false><<<gemmGridD, 256, 0, stream>>>(
      Axn, pwv, bv, nullptr, (void*)Vt, Dm, Dm);  // scatter V^T per head

  // ---- TE injection on q,k ----
  te_inject<<<(unsigned)(MD / 256), 256, 0, stream>>>(Bq, Dk, TE, wq_ht, bq_ht,
                                                      wk_ht, bk_ht);

  // ---- two conv refinement passes (ping-pong) ----
  conv_gemm<<<convGrid, 256, 0, stream>>>(Bq, pc1, c1b, Cq);
  conv_gemm<<<convGrid, 256, 0, stream>>>(Dk, pc1, c1b, Ek);
  conv_gemm<<<convGrid, 256, 0, stream>>>(Cq, pc2, c2b, Bq);
  conv_gemm<<<convGrid, 256, 0, stream>>>(Ek, pc2, c2b, Dk);

  // ---- attention: 8192 heads, 4 waves/block ----
  attention_kernel<<<(Bsz * Nsz * HEADS) / 4, 128, 0, stream>>>(Bq, Dk, Vt, G);

  // ---- output projection + residual: x1 = x + G@wo + bo (f32) ----
  gemm_bf16<1, false, true><<<gemmGridD, 256, 0, stream>>>(
      G, pwo, bo, x, (void*)H, Dm, Dm);

  // ---- LN2: x1 -> h (bf16, reuse Axn) ----
  layernorm_bf16<<<MTOK, 256, 0, stream>>>(H, g2, be2, Axn);

  // ---- FFN1 (ReLU) -> hidden bf16 [M x 2048] ----
  gemm_bf16<0, true, false><<<gemmGridF1, 256, 0, stream>>>(
      Axn, pf1, bf1, nullptr, (void*)hid, Dm, 4 * Dm);

  // ---- FFN2 + residual(x1) -> d_out f32 ----
  gemm_bf16<1, false, true><<<gemmGridD, 256, 0, stream>>>(
      hid, pf2, bf2, H, d_out, 4 * Dm, Dm);
}
